// AdvancedFRBDetector_13537736917694
// MI455X (gfx1250) — matrix operand order repaired and checked by reading, not compiled, not run on hardware
//
#include <hip/hip_runtime.h>
#include <hip/hip_bf16.h>

// ---------------------------------------------------------------------------
// MI455X (gfx1250) implementation of AdvancedFRBDetector forward pass.
// wave32; all GEMM-shaped work (dense layers AND convs via implicit GEMM)
// runs on v_wmma_f32_16x16x32_bf16 with f32 accumulation.
// Dense GEMM is register-staged + LDS double-buffered (1 barrier/k-tile).
// ---------------------------------------------------------------------------

typedef __bf16  bf16x16 __attribute__((ext_vector_type(16)));
typedef float   floatx8 __attribute__((ext_vector_type(8)));

#define BN_SCALE 0.9999950000374997f   // 1/sqrt(1+1e-5)

__device__ __forceinline__ unsigned short f2bf(float x) {
    unsigned int u = __float_as_uint(x);
    u += 0x7FFFu + ((u >> 16) & 1u);   // round-to-nearest-even
    return (unsigned short)(u >> 16);
}
// pack two f32 -> packed bf16 dword (v_cvt_pk_bf16_f32 when available)
__device__ __forceinline__ unsigned int pack_bf16(float x, float y) {
#if __has_builtin(__builtin_amdgcn_cvt_pk_bf16_f32)
    typedef __bf16 bf16x2 __attribute__((ext_vector_type(2)));
    bf16x2 r = __builtin_amdgcn_cvt_pk_bf16_f32(x, y);
    return __builtin_bit_cast(unsigned int, r);
#else
    return (unsigned)f2bf(x) | ((unsigned)f2bf(y) << 16);
#endif
}
__device__ __forceinline__ float sigmoidf(float x) { return 1.0f / (1.0f + __expf(-x)); }

#define GBM 64
#define GBN 64
#define GBK 32
#define GLDA 34   // u16 stride (even -> 4B-aligned K-pair reads)
#define GLDB 34

// ---- shared WMMA tile core: fragments built per ISA VGPR layouts ----
struct WmmaCtx { int lane, wave, wm, wn, hi, lm; };

__device__ __forceinline__ void wmma_step(const unsigned short* As, const unsigned short* Bs,
                                          const WmmaCtx& c, floatx8 acc[2][2])
{
    bf16x16 afr[2], bfr[2];
    #pragma unroll
    for (int i = 0; i < 2; ++i) {
        union { bf16x16 v; unsigned int u[8]; } ua;
        const unsigned short* ap = &As[(c.wm + 16 * i + c.lm) * GLDA];
        #pragma unroll
        for (int v8 = 0; v8 < 8; ++v8) {
            int kb = ((v8 < 4) ? 0 : 16) + (c.hi ? 8 : 0) + (v8 & 3) * 2;
            ua.u[v8] = *(const unsigned int*)(ap + kb);
        }
        afr[i] = ua.v;
        union { bf16x16 v; unsigned int u[8]; } ub;
        const unsigned short* bp = &Bs[(c.wn + 16 * i + c.lm) * GLDB];
        #pragma unroll
        for (int v8 = 0; v8 < 8; ++v8) {
            int kb = ((v8 < 4) ? 0 : 16) + (c.hi ? 8 : 0) + (v8 & 3) * 2;
            ub.u[v8] = *(const unsigned int*)(bp + kb);
        }
        bfr[i] = ub.v;
    }
    #pragma unroll
    for (int i = 0; i < 2; ++i)
        #pragma unroll
        for (int j = 0; j < 2; ++j)
            acc[i][j] = __builtin_amdgcn_wmma_f32_16x16x32_bf16(
                false, afr[i], false, bfr[j], (short)0, acc[i][j], false, false);
}

// ============================================================================
// Generic bf16 WMMA GEMM: out[M,N] = epilogue(A[M,K] @ W + bias)
// REQUIREMENTS (hold for every call in this net): K % 32 == 0, M % 64 == 0,
// N % 4 == 0.  wTrans==0: W is [K,N]; wTrans==1: W is [N,K].
// act: 0=none 1=relu 2=swish.  out = resid + alpha*act(acc+bias).
// Register-staged, LDS double-buffered pipeline.
// ============================================================================
__device__ __forceinline__ void g_load(const float* __restrict__ A, const float* __restrict__ W,
                                       int N, int K, int m0, int n0, int k0, int tid, int wTrans,
                                       float4 av[4], float4 bv[4])
{
    #pragma unroll
    for (int it = 0; it < 4; ++it) {
        int i = tid + it * 128;
        int m = i >> 3, kq = (i & 7) * 4;
        av[it] = *(const float4*)(A + (size_t)(m0 + m) * K + (k0 + kq));
    }
    if (wTrans) {   // W[N,K], contiguous along k
        #pragma unroll
        for (int it = 0; it < 4; ++it) {
            int i = tid + it * 128;
            int n = i >> 3, kq = (i & 7) * 4;
            int gn = n0 + n;
            bv[it] = (gn < N) ? *(const float4*)(W + (size_t)gn * K + (k0 + kq))
                              : make_float4(0.f, 0.f, 0.f, 0.f);
        }
    } else {        // W[K,N], contiguous along n (N % 4 == 0 -> group-uniform guard)
        #pragma unroll
        for (int it = 0; it < 4; ++it) {
            int i = tid + it * 128;
            int k = i >> 4, nq = (i & 15) * 4;
            int gn = n0 + nq;
            bv[it] = (gn < N) ? *(const float4*)(W + (size_t)(k0 + k) * N + gn)
                              : make_float4(0.f, 0.f, 0.f, 0.f);
        }
    }
}

__device__ __forceinline__ void g_store(unsigned short* As, unsigned short* Bs,
                                        int tid, int wTrans,
                                        const float4 av[4], const float4 bv[4])
{
    #pragma unroll
    for (int it = 0; it < 4; ++it) {
        int i = tid + it * 128;
        int m = i >> 3, kq = (i & 7) * 4;
        unsigned int* dst = (unsigned int*)&As[m * GLDA + kq];
        dst[0] = pack_bf16(av[it].x, av[it].y);
        dst[1] = pack_bf16(av[it].z, av[it].w);
    }
    if (wTrans) {
        #pragma unroll
        for (int it = 0; it < 4; ++it) {
            int i = tid + it * 128;
            int n = i >> 3, kq = (i & 7) * 4;
            unsigned int* dst = (unsigned int*)&Bs[n * GLDB + kq];
            dst[0] = pack_bf16(bv[it].x, bv[it].y);
            dst[1] = pack_bf16(bv[it].z, bv[it].w);
        }
    } else {        // transpose into Bs[n][k]
        #pragma unroll
        for (int it = 0; it < 4; ++it) {
            int i = tid + it * 128;
            int k = i >> 4, nq = (i & 15) * 4;
            Bs[(nq + 0) * GLDB + k] = f2bf(bv[it].x);
            Bs[(nq + 1) * GLDB + k] = f2bf(bv[it].y);
            Bs[(nq + 2) * GLDB + k] = f2bf(bv[it].z);
            Bs[(nq + 3) * GLDB + k] = f2bf(bv[it].w);
        }
    }
}

__global__ __launch_bounds__(128) void k_gemm_bf16(
    const float* __restrict__ A, const float* __restrict__ W,
    const float* __restrict__ bias, const float* __restrict__ resid,
    float* __restrict__ out, int M, int N, int K, int act, float alpha, int wTrans)
{
    __shared__ unsigned short As[2][GBM * GLDA];
    __shared__ unsigned short Bs[2][GBN * GLDB];

    const int m0 = blockIdx.y * GBM;
    const int n0 = blockIdx.x * GBN;
    const int tid = threadIdx.x;
    WmmaCtx c;
    c.lane = tid & 31; c.wave = tid >> 5;
    c.wm = (c.wave & 1) * 32; c.wn = (c.wave >> 1) * 32;
    c.hi = c.lane >> 4; c.lm = c.lane & 15;

    floatx8 acc[2][2] = {};
    float4 av[4], bv[4];

    const int nk = K / GBK;
    g_load(A, W, N, K, m0, n0, 0, tid, wTrans, av, bv);
    g_store(As[0], Bs[0], tid, wTrans, av, bv);
    __syncthreads();

    for (int kt = 0; kt < nk; ++kt) {
        int cur = kt & 1;
        if (kt + 1 < nk) {
            g_load(A, W, N, K, m0, n0, (kt + 1) * GBK, tid, wTrans, av, bv);
            if (kt + 2 < nk)   // pull k-tile kt+2 toward L2 (global_prefetch path)
                __builtin_prefetch(A + (size_t)(m0 + (tid & 63)) * K + (kt + 2) * GBK, 0, 1);
        }
        wmma_step(As[cur], Bs[cur], c, acc);       // overlaps in-flight b128 loads
        if (kt + 1 < nk) g_store(As[cur ^ 1], Bs[cur ^ 1], tid, wTrans, av, bv);
        __syncthreads();
    }

    // ---- epilogue ----
    #pragma unroll
    for (int i = 0; i < 2; ++i) {
        #pragma unroll
        for (int j = 0; j < 2; ++j) {
            #pragma unroll
            for (int r = 0; r < 8; ++r) {
                int row = m0 + c.wm + 16 * i + r + (c.hi ? 8 : 0);
                int col = n0 + c.wn + 16 * j + c.lm;
                if (row < M && col < N) {
                    float v = acc[i][j][r];
                    if (bias) v += bias[col];
                    if (act == 1)       v = fmaxf(v, 0.0f);
                    else if (act == 2)  v = v * sigmoidf(v);
                    v *= alpha;
                    if (resid) v += resid[(size_t)row * N + col];
                    out[(size_t)row * N + col] = v;
                }
            }
        }
    }
}

// ============================================================================
// Implicit-GEMM conv2d (NCHW, OIHW) fused with bias + eval-BN + ReLU, on WMMA.
//   M = B*HO*WO (wo fastest), N = Co, K = Ci*KH*KW (kw fastest).
//   grid.x = M tiles, grid.y = N tiles.  (A staging is a data-dependent gather.)
// ============================================================================
__global__ __launch_bounds__(128) void k_conv_gemm_bf16(
    const float* __restrict__ in, const float* __restrict__ w,
    const float* __restrict__ cb, const float* __restrict__ g,
    const float* __restrict__ bb, float* __restrict__ out,
    int B_, int Ci, int H, int W_, int Co, int KH, int KW,
    int SH, int SW, int PH, int PW, int HO, int WO)
{
    __shared__ unsigned short As[GBM * GLDA];
    __shared__ unsigned short Bs[GBN * GLDB];

    const int M = B_ * HO * WO, N = Co, K = Ci * KH * KW;
    const int m0 = blockIdx.x * GBM;
    const int n0 = blockIdx.y * GBN;
    const int tid = threadIdx.x;
    WmmaCtx c;
    c.lane = tid & 31; c.wave = tid >> 5;
    c.wm = (c.wave & 1) * 32; c.wn = (c.wave >> 1) * 32;
    c.hi = c.lane >> 4; c.lm = c.lane & 15;

    floatx8 acc[2][2] = {};

    for (int k0 = 0; k0 < K; k0 += GBK) {
        // ---- A stage: im2col gather ----
        for (int i = tid; i < GBM * GBK; i += 128) {
            int mm = i >> 5, kk = i & 31;
            int gm = m0 + mm, gk = k0 + kk;
            float v = 0.0f;
            if (gm < M && gk < K) {
                int wo = gm % WO; int t1 = gm / WO;
                int ho = t1 % HO; int b  = t1 / HO;
                int kw = gk % KW; int t2 = gk / KW;
                int kh = t2 % KH; int ci = t2 / KH;
                int hin = ho * SH + kh - PH;
                int win = wo * SW + kw - PW;
                if (hin >= 0 && hin < H && win >= 0 && win < W_)
                    v = in[((size_t)(b * Ci + ci) * H + hin) * W_ + win];
            }
            As[mm * GLDA + kk] = f2bf(v);
        }
        // ---- B stage: w is [Co, K] contiguous along k ----
        if ((K & 3) == 0) {   // rows 16B-reachable via dword-aligned b128 groups
            #pragma unroll
            for (int it = 0; it < 4; ++it) {
                int i = tid + it * 128;
                int n = i >> 3, kq = (i & 7) * 4;
                int gn = n0 + n, gk = k0 + kq;
                float4 v = make_float4(0.f, 0.f, 0.f, 0.f);
                if (gn < N && gk < K)       // K%4==0 -> group fully valid
                    v = *(const float4*)(w + (size_t)gn * K + gk);
                unsigned int* dst = (unsigned int*)&Bs[n * GLDB + kq];
                dst[0] = pack_bf16(v.x, v.y);
                dst[1] = pack_bf16(v.z, v.w);
            }
        } else {              // K=9/21 (Ci==1 convs): scalar staging, any alignment
            for (int i = tid; i < GBN * GBK; i += 128) {
                int n = i >> 5, kk = i & 31;
                int gn = n0 + n, gk = k0 + kk;
                float v = (gn < N && gk < K) ? w[(size_t)gn * K + gk] : 0.0f;
                Bs[n * GLDB + kk] = f2bf(v);
            }
        }
        __syncthreads();
        wmma_step(As, Bs, c, acc);
        __syncthreads();
    }

    // ---- epilogue: bias + BN + ReLU, scatter to NCHW ----
    #pragma unroll
    for (int i = 0; i < 2; ++i) {
        #pragma unroll
        for (int j = 0; j < 2; ++j) {
            #pragma unroll
            for (int r = 0; r < 8; ++r) {
                int row = m0 + c.wm + 16 * i + r + (c.hi ? 8 : 0);
                int col = n0 + c.wn + 16 * j + c.lm;
                if (row < M && col < N) {
                    float v = acc[i][j][r];
                    v = (v + cb[col]) * (BN_SCALE * g[col]) + bb[col];
                    v = fmaxf(v, 0.0f);
                    int wo = row % WO; int t1 = row / WO;
                    int ho = t1 % HO; int b  = t1 / HO;
                    out[((size_t)(b * Co + col) * HO + ho) * WO + wo] = v;
                }
            }
        }
    }
}

// ============================================================================
// Adaptive average pool (evenly divisible)
// ============================================================================
__global__ void k_pool(const float* __restrict__ in, float* __restrict__ out,
                       int N, int C, int H, int W, int OH, int OW)
{
    int idx = blockIdx.x * 256 + threadIdx.x;
    int total = N * C * OH * OW;
    if (idx >= total) return;
    int ow = idx % OW; int t = idx / OW;
    int oh = t % OH;   t /= OH;
    int cc = t % C;    int n = t / C;
    int bh = H / OH, bw = W / OW;
    const float* p = in + (size_t)((n * C + cc) * H + oh * bh) * W + (size_t)ow * bw;
    float s = 0.0f;
    for (int i = 0; i < bh; ++i)
        for (int j = 0; j < bw; ++j)
            s += p[(size_t)i * W + j];
    out[idx] = s / (float)(bh * bw);
}

// ============================================================================
// Small dense layer (one thread per output element)
// ============================================================================
__global__ void k_linear_small(const float* __restrict__ in, const float* __restrict__ w,
                               const float* __restrict__ b, float* __restrict__ out,
                               int M, int K, int N, int act)
{
    int idx = blockIdx.x * 256 + threadIdx.x;
    if (idx >= M * N) return;
    int m = idx / N, n = idx % N;
    float s = b ? b[n] : 0.0f;
    const float* ip = in + (size_t)m * K;
    for (int k = 0; k < K; ++k) s = fmaf(ip[k], w[(size_t)k * N + n], s);
    if (act == 1) s = fmaxf(s, 0.0f);
    out[idx] = s;
}

// ============================================================================
// DM head: sigmoid-scale, sort(16), per-chunk integer dedispersion shifts
// ============================================================================
__global__ void k_dm_shift(const float* __restrict__ l3out,
                           float* __restrict__ dm, int* __restrict__ shift)
{
    int b = threadIdx.x;
    if (b >= 8) return;
    float v[16];
    for (int d = 0; d < 16; ++d)
        v[d] = 0.1f + sigmoidf(l3out[b * 16 + d]) * 0.9f;
    for (int i = 1; i < 16; ++i) {            // insertion sort ascending
        float key = v[i]; int j = i - 1;
        while (j >= 0 && v[j] > key) { v[j + 1] = v[j]; --j; }
        v[j + 1] = key;
    }
    for (int d = 0; d < 16; ++d) dm[b * 16 + d] = v[d];

    const float dfmax = 1.0f - 1.0f / 2.25f;  // df at f=0 (fmax=1.5 GHz)
    for (int cc = 0; cc < 8; ++cc) {
        float md = 0.0f;
        for (int j = 0; j < 128; ++j) {
            int jj = cc * 128 + j;
            float fi = (float)jj / 1023.0f;
            float fg = 1.0f + 0.5f * fi;
            float df = 1.0f / (fg * fg) - 1.0f / 2.25f;
            md += df / (dfmax + 1e-8f) * 102.4f;   // T*0.2
        }
        md *= (1.0f / 128.0f);
        for (int d = 0; d < 16; ++d)
            shift[(b * 16 + d) * 8 + cc] = (int)(v[d] * md);   // trunc == astype(int32)
    }
}

// ============================================================================
// Fused dedispersion gather + 32x32 adaptive pool -> pooled[128,1024]
// (x is 16 MB -> L2-resident; all 16 DM-trial gathers hit L2)
// ============================================================================
__global__ void k_pool_dedisp(const float* __restrict__ x, const int* __restrict__ shift,
                              float* __restrict__ pooled)
{
    int idx = blockIdx.x * 256 + threadIdx.x;     // total 128*1024
    if (idx >= 128 * 1024) return;
    int ow = idx & 31;
    int oh = (idx >> 5) & 31;
    int bd = idx >> 10;
    int b  = bd >> 4;
    float s = 0.0f;
    for (int ti = 0; ti < 16; ++ti) {
        int t = oh * 16 + ti;
        for (int fj = 0; fj < 32; ++fj) {
            int f  = ow * 32 + fj;
            int cc = f >> 7;
            int ts = (t + shift[bd * 8 + cc]) & 511;
            s += x[((size_t)b * 512 + ts) * 1024 + f];
        }
    }
    pooled[idx] = s * (1.0f / 512.0f);
}

// ============================================================================
// Attention-weighted dedispersed sum -> dmc[8,512,1024]
// ============================================================================
__global__ void k_dmc(const float* __restrict__ x, const int* __restrict__ shift,
                      const float* __restrict__ aw, float* __restrict__ dmc)
{
    int idx = blockIdx.x * 256 + threadIdx.x;     // total 8*512*1024
    if (idx >= 8 * 512 * 1024) return;
    int f = idx & 1023;
    int t = (idx >> 10) & 511;
    int b = idx >> 19;
    int cc = f >> 7;
    float s = 0.0f;
    for (int d = 0; d < 16; ++d) {
        int ts = (t + shift[(b * 16 + d) * 8 + cc]) & 511;
        s = fmaf(aw[b * 16 + d], x[((size_t)b * 512 + ts) * 1024 + f], s);
    }
    dmc[idx] = s;
}

__global__ void k_softmax16(const float* __restrict__ in, float* __restrict__ out)
{
    int b = threadIdx.x;
    if (b >= 8) return;
    float mx = -1e30f;
    for (int d = 0; d < 16; ++d) mx = fmaxf(mx, in[b * 16 + d]);
    float e[16], s = 0.0f;
    for (int d = 0; d < 16; ++d) { e[d] = __expf(in[b * 16 + d] - mx); s += e[d]; }
    float inv = 1.0f / s;
    for (int d = 0; d < 16; ++d) out[b * 16 + d] = e[d] * inv;
}

// [8,256,32,16] (NCHW) -> proj input rows (b*32+t), cols c*16+f
__global__ void k_transpose_proj(const float* __restrict__ h, float* __restrict__ out)
{
    int idx = blockIdx.x * 256 + threadIdx.x;     // total 1,048,576
    if (idx >= (1 << 20)) return;
    int f = idx & 15;
    int t = (idx >> 4) & 31;
    int cc = (idx >> 9) & 255;
    int b = idx >> 17;
    out[((size_t)(b * 32 + t) << 12) + (cc << 4) + f] = h[idx];
}

// LayerNorm over 256 cols; one wave32 per row, shuffle reduction
__global__ void k_layernorm(const float* __restrict__ in, const float* __restrict__ g,
                            const float* __restrict__ b, float* __restrict__ out, int cols)
{
    int row = blockIdx.x, lane = threadIdx.x;
    const float* p = in + (size_t)row * cols;
    float s = 0.0f, s2 = 0.0f;
    for (int cc = lane; cc < cols; cc += 32) { float v = p[cc]; s += v; s2 += v * v; }
    for (int o = 16; o; o >>= 1) { s += __shfl_down(s, o, 32); s2 += __shfl_down(s2, o, 32); }
    s = __shfl(s, 0, 32); s2 = __shfl(s2, 0, 32);
    float mean = s / cols;
    float var  = s2 / cols - mean * mean;
    float inv  = rsqrtf(var + 1e-5f);
    float* q = out + (size_t)row * cols;
    for (int cc = lane; cc < cols; cc += 32)
        q[cc] = (p[cc] - mean) * inv * g[cc] + b[cc];
}

// Self-attention, one block per (batch, head); T=32, Dh=64
__global__ void k_attention(const float* __restrict__ qkv, float* __restrict__ o)
{
    int bh = blockIdx.x;
    int b = bh >> 2, h = bh & 3;
    int i = threadIdx.x;                  // 32 threads = query rows
    __shared__ float Ks[32][64], Vs[32][64];
    const float* base = qkv + (size_t)(b * 32 + i) * 768;
    for (int d = 0; d < 64; ++d) {
        Ks[i][d] = base[256 + h * 64 + d];
        Vs[i][d] = base[512 + h * 64 + d];
    }
    __syncthreads();
    float q[64];
    for (int d = 0; d < 64; ++d) q[d] = base[h * 64 + d];
    float sc[32], mx = -1e30f;
    for (int j = 0; j < 32; ++j) {
        float s = 0.0f;
        for (int d = 0; d < 64; ++d) s = fmaf(q[d], Ks[j][d], s);
        s *= 0.125f;                      // Dh^-0.5
        sc[j] = s; mx = fmaxf(mx, s);
    }
    float sum = 0.0f;
    for (int j = 0; j < 32; ++j) { sc[j] = __expf(sc[j] - mx); sum += sc[j]; }
    float inv = 1.0f / sum;
    float* op = o + (size_t)(b * 32 + i) * 256 + h * 64;
    for (int d = 0; d < 64; ++d) {
        float a = 0.0f;
        for (int j = 0; j < 32; ++j) a = fmaf(sc[j], Vs[j][d], a);
        op[d] = a * inv;
    }
}

__global__ void k_glu(const float* __restrict__ pw1, float* __restrict__ out)
{
    int idx = blockIdx.x * 256 + threadIdx.x;     // 256 rows x 256 cols
    if (idx >= 65536) return;
    int cc = idx & 255, m = idx >> 8;
    float a = pw1[(size_t)m * 512 + cc];
    float gt = pw1[(size_t)m * 512 + 256 + cc];
    out[idx] = a * sigmoidf(gt);
}

// Depthwise conv1d k=15 pad=7 along T + eval-BN + swish
__global__ void k_dwconv(const float* __restrict__ in, const float* __restrict__ dww,
                         const float* __restrict__ dwb, const float* __restrict__ bng,
                         const float* __restrict__ bnb, float* __restrict__ out)
{
    int idx = blockIdx.x * 256 + threadIdx.x;     // total 65536
    if (idx >= 65536) return;
    int cc = idx & 255;
    int t = (idx >> 8) & 31;
    int b = idx >> 13;
    float s = dwb[cc];
    for (int j = 0; j < 15; ++j) {
        int tt = t + j - 7;
        if (tt < 0 || tt >= 32) continue;
        s = fmaf(in[((size_t)(b * 32 + tt) << 8) + cc], dww[cc * 15 + j], s);
    }
    s = s * (BN_SCALE * bng[cc]) + bnb[cc];
    out[idx] = s * sigmoidf(s);
}

__global__ void k_rowmean(const float* __restrict__ feat, float* __restrict__ gf)
{
    int idx = blockIdx.x * 256 + threadIdx.x;     // total 8*256
    if (idx >= 2048) return;
    int cc = idx & 255, b = idx >> 8;
    float s = 0.0f;
    for (int t = 0; t < 32; ++t) s += feat[((size_t)(b * 32 + t) << 8) + cc];
    gf[idx] = s * (1.0f / 32.0f);
}

__global__ void k_comb(const float* __restrict__ gf, const float* __restrict__ aw,
                       float* __restrict__ comb)
{
    int idx = blockIdx.x * 256 + threadIdx.x;     // total 8*272
    if (idx >= 8 * 272) return;
    int b = idx / 272, j = idx % 272;
    comb[idx] = (j < 256) ? gf[b * 256 + j] : aw[b * 16 + (j - 256)];
}

// ============================================================================
// Host orchestration
// ============================================================================
static inline int cdiv(int a, int b) { return (a + b - 1) / b; }

struct ConfP {
    const float *ln1g,*ln1b,*ff1aw,*ff1ab,*ff1bw,*ff1bb,*ln2g,*ln2b,*qkvw,*oww,*owb,
                *ln3g,*ln3b,*pw1w,*pw1b,*dww,*dwb,*bng,*bnb,*pw2w,*pw2b,
                *ln4g,*ln4b,*ff2aw,*ff2ab,*ff2bw,*ff2bb,*fng,*fnb;
};

extern "C" void kernel_launch(void* const* d_in, const int* in_sizes, int n_in,
                              void* d_out, int out_size, void* d_ws, size_t ws_size,
                              hipStream_t stream)
{
    (void)in_sizes; (void)n_in; (void)out_size; (void)ws_size;

    // ---- input leaves in setup_inputs() insertion order ----
    int p = 0;
    auto F = [&](void) { return (const float*)d_in[p++]; };
    const float* x   = F();
    const float *c1w=F(),*c1b=F(),*g1=F(),*b1=F();
    const float *c2w=F(),*c2b=F(),*g2=F(),*b2=F();
    const float *l1w=F(),*l1b=F(),*l2w=F(),*l2b=F(),*l3w=F(),*l3b=F();
    const float *f1w=F(),*f1b=F(),*f2w=F(),*f2b=F();
    const float *a1w=F(),*a1b=F(),*a2w=F(),*a2b=F(),*a3w=F(),*a3b=F();
    struct { const float *w,*b,*g,*be; } fcp[4];
    for (int i = 0; i < 4; ++i) { fcp[i].w=F(); fcp[i].b=F(); fcp[i].g=F(); fcp[i].be=F(); }
    const float *projw=F(), *projb=F();
    ConfP cf[3];
    for (int i = 0; i < 3; ++i) {
        ConfP& q = cf[i];
        q.ln1g=F(); q.ln1b=F(); q.ff1aw=F(); q.ff1ab=F(); q.ff1bw=F(); q.ff1bb=F();
        q.ln2g=F(); q.ln2b=F(); q.qkvw=F();  q.oww=F();  q.owb=F();
        q.ln3g=F(); q.ln3b=F(); q.pw1w=F();  q.pw1b=F(); q.dww=F();  q.dwb=F();
        q.bng=F();  q.bnb=F();  q.pw2w=F();  q.pw2b=F();
        q.ln4g=F(); q.ln4b=F(); q.ff2aw=F(); q.ff2ab=F(); q.ff2bw=F(); q.ff2bb=F();
        q.fng=F();  q.fnb=F();
    }
    const float *cl1w=F(), *cl1b=F(), *cl2w=F(), *cl2b=F();

    // ---- workspace arena (floats); R0/R1 ping-pong for big activations ----
    float* ws = (float*)d_ws;
    const size_t R0 = 0;                        // 33,554,432 floats (134 MB)
    const size_t R1 = 33554432;                 // 16,777,216 floats ( 67 MB)
    size_t o = R1 + 16777216;                   // small-buffer region
    float* pool1  = ws + o; o += 8 * 4096;
    float* mlp1   = ws + o; o += 8 * 256;
    float* mlp2   = ws + o; o += 8 * 128;
    float* l3out  = ws + o; o += 8 * 16;
    float* dm     = ws + o; o += 8 * 16;
    int*   shift  = (int*)(ws + o); o += 8 * 16 * 8;
    float* pooled = ws + o; o += 128 * 1024;
    float* f1out  = ws + o; o += 128 * 128;
    float* ft     = ws + o; o += 128 * 32;      // == [8,512] flat
    float* a1out  = ws + o; o += 8 * 256;
    float* a2out  = ws + o; o += 8 * 64;
    float* a3out  = ws + o; o += 8 * 16;
    float* aw     = ws + o; o += 8 * 16;
    float* featA  = ws + o; o += 256 * 256;
    float* featB  = ws + o; o += 256 * 256;
    float* t_ln   = ws + o; o += 256 * 256;
    float* t_ff1  = ws + o; o += 256 * 1024;
    float* qkvbuf = ws + o; o += 256 * 768;
    float* obuf   = ws + o; o += 256 * 256;
    float* pw1out = ws + o; o += 256 * 512;
    float* gluout = ws + o; o += 256 * 256;
    float* dwout  = ws + o; o += 256 * 256;
    float* gf     = ws + o; o += 8 * 256;
    float* comb   = ws + o; o += 8 * 272;
    float* cls1o  = ws + o; o += 8 * 128;

    float* h1    = ws + R1;                     // [8,16,512,256]
    float* h2    = ws + R0;                     // [8,32,256,64]
    float* dmc   = ws + R1;                     // [8,512,1024]   (h1 dead)
    float* fco1  = ws + R0;                     // [8,32,256,512] (h2 dead)
    float* fco2  = ws + R1;                     // [8,64,128,256] (dmc dead)
    float* fco3  = ws + R0;                     // [8,128,64,64]  (fco1 dead)
    float* fco4  = ws + R1;                     // [8,256,32,16]  (fco2 dead)
    float* projin= ws + R1 + 1048576;           // [256,4096]

    auto gemm = [&](const float* A, const float* W, const float* bias, const float* resid,
                    float* outp, int M, int N, int K, int act, float alpha, int wT) {
        dim3 g(cdiv(N, GBN), cdiv(M, GBM));
        k_gemm_bf16<<<g, 128, 0, stream>>>(A, W, bias, resid, outp, M, N, K, act, alpha, wT);
    };
    auto conv = [&](const float* in, const float* w, const float* cb, const float* g,
                    const float* bb, float* outp, int B_, int Ci, int H, int W_,
                    int Co, int KH, int KW, int SH, int SW, int PH, int PW, int HO, int WO) {
        int M = B_ * HO * WO;
        dim3 gg(cdiv(M, GBM), cdiv(Co, GBN));
        k_conv_gemm_bf16<<<gg, 128, 0, stream>>>(
            in, w, cb, g, bb, outp, B_, Ci, H, W_, Co, KH, KW, SH, SW, PH, PW, HO, WO);
    };

    // ---------------- DynamicDMPredictor ----------------
    conv(x,  c1w, c1b, g1, b1, h1, 8, 1, 512, 1024, 16, 3, 7, 1, 4, 1, 3, 512, 256);
    conv(h1, c2w, c2b, g2, b2, h2, 8, 16, 512, 256, 32, 3, 5, 2, 4, 1, 2, 256, 64);
    k_pool<<<cdiv(8 * 32 * 8 * 16, 256), 256, 0, stream>>>(h2, pool1, 8, 32, 256, 64, 8, 16);
    k_linear_small<<<cdiv(8 * 256, 256), 256, 0, stream>>>(pool1, l1w, l1b, mlp1, 8, 4096, 256, 1);
    k_linear_small<<<cdiv(8 * 128, 256), 256, 0, stream>>>(mlp1,  l2w, l2b, mlp2, 8, 256, 128, 1);
    k_linear_small<<<1, 256, 0, stream>>>(mlp2, l3w, l3b, l3out, 8, 128, 16, 0);
    k_dm_shift<<<1, 8, 0, stream>>>(l3out, dm, shift);

    // ---------------- DMAttentionModule ----------------
    k_pool_dedisp<<<cdiv(128 * 1024, 256), 256, 0, stream>>>(x, shift, pooled);
    gemm(pooled, f1w, f1b, nullptr, f1out, 128, 128, 1024, 1, 1.0f, 0);
    gemm(f1out,  f2w, f2b, nullptr, ft,    128, 32,  128,  1, 1.0f, 0);
    k_linear_small<<<cdiv(8 * 256, 256), 256, 0, stream>>>(ft,    a1w, a1b, a1out, 8, 512, 256, 1);
    k_linear_small<<<cdiv(8 * 64, 256), 256, 0, stream>>>(a1out, a2w, a2b, a2out, 8, 256, 64, 1);
    k_linear_small<<<1, 256, 0, stream>>>(a2out, a3w, a3b, a3out, 8, 64, 16, 0);
    k_softmax16<<<1, 8, 0, stream>>>(a3out, aw);
    k_dmc<<<cdiv(8 * 512 * 1024, 256), 256, 0, stream>>>(x, shift, aw, dmc);

    // ---------------- freq_compress ----------------
    conv(dmc,  fcp[0].w, fcp[0].b, fcp[0].g, fcp[0].be, fco1, 8, 1,   512, 1024, 32,  3, 3, 2, 2, 1, 1, 256, 512);
    conv(fco1, fcp[1].w, fcp[1].b, fcp[1].g, fcp[1].be, fco2, 8, 32,  256, 512,  64,  3, 3, 2, 2, 1, 1, 128, 256);
    conv(fco2, fcp[2].w, fcp[2].b, fcp[2].g, fcp[2].be, fco3, 8, 64,  128, 256,  128, 3, 3, 2, 4, 1, 1, 64,  64);
    conv(fco3, fcp[3].w, fcp[3].b, fcp[3].g, fcp[3].be, fco4, 8, 128, 64,  64,   256, 3, 3, 2, 4, 1, 1, 32,  16);
    k_transpose_proj<<<cdiv(1 << 20, 256), 256, 0, stream>>>(fco4, projin);
    gemm(projin, projw, projb, nullptr, featA, 256, 256, 4096, 0, 1.0f, 0);

    // ---------------- Conformer stack ----------------
    float* feat  = featA;
    float* featN = featB;
    for (int blk = 0; blk < 3; ++blk) {
        const ConfP& q = cf[blk];
        // FFN-1 (half-step residual)
        k_layernorm<<<256, 32, 0, stream>>>(feat, q.ln1g, q.ln1b, t_ln, 256);
        gemm(t_ln,  q.ff1aw, q.ff1ab, nullptr, t_ff1, 256, 1024, 256, 2, 1.0f, 0);
        gemm(t_ff1, q.ff1bw, q.ff1bb, feat,    feat,  256, 256, 1024, 0, 0.5f, 0);
        // MHSA
        k_layernorm<<<256, 32, 0, stream>>>(feat, q.ln2g, q.ln2b, t_ln, 256);
        gemm(t_ln, q.qkvw, nullptr, nullptr, qkvbuf, 256, 768, 256, 0, 1.0f, 0);
        k_attention<<<32, 32, 0, stream>>>(qkvbuf, obuf);
        gemm(obuf, q.oww, q.owb, feat, feat, 256, 256, 256, 0, 1.0f, 0);
        // Conv module
        k_layernorm<<<256, 32, 0, stream>>>(feat, q.ln3g, q.ln3b, t_ln, 256);
        gemm(t_ln, q.pw1w, q.pw1b, nullptr, pw1out, 256, 512, 256, 0, 1.0f, 1);
        k_glu<<<cdiv(65536, 256), 256, 0, stream>>>(pw1out, gluout);
        k_dwconv<<<cdiv(65536, 256), 256, 0, stream>>>(gluout, q.dww, q.dwb, q.bng, q.bnb, dwout);
        gemm(dwout, q.pw2w, q.pw2b, feat, feat, 256, 256, 256, 0, 1.0f, 1);
        // FFN-2 (half-step residual)
        k_layernorm<<<256, 32, 0, stream>>>(feat, q.ln4g, q.ln4b, t_ln, 256);
        gemm(t_ln,  q.ff2aw, q.ff2ab, nullptr, t_ff1, 256, 1024, 256, 2, 1.0f, 0);
        gemm(t_ff1, q.ff2bw, q.ff2bb, feat,    feat,  256, 256, 1024, 0, 0.5f, 0);
        // Final LN -> swap buffers
        k_layernorm<<<256, 32, 0, stream>>>(feat, q.fng, q.fnb, featN, 256);
        float* tmp = feat; feat = featN; featN = tmp;
    }

    // ---------------- Classifier ----------------
    k_rowmean<<<cdiv(2048, 256), 256, 0, stream>>>(feat, gf);
    k_comb<<<cdiv(8 * 272, 256), 256, 0, stream>>>(gf, aw, comb);
    k_linear_small<<<cdiv(8 * 128, 256), 256, 0, stream>>>(comb, cl1w, cl1b, cls1o, 8, 272, 128, 1);
    k_linear_small<<<1, 256, 0, stream>>>(cls1o, cl2w, cl2b, (float*)d_out, 8, 128, 2, 0);
}
